// ExportableShiftedWindowAttention_3161095929996
// MI455X (gfx1250) — matrix operand order repaired and checked by reading, not compiled, not run on hardware
//
#include <hip/hip_runtime.h>

typedef __attribute__((ext_vector_type(16))) _Float16 v16h;
typedef __attribute__((ext_vector_type(8)))  _Float16 h8;
typedef __attribute__((ext_vector_type(4)))  _Float16 h4;
typedef __attribute__((ext_vector_type(8)))  float    v8f;

#define CC 128
#define NTOK 49
#define NPAD 64

// LDS leading dims (in halves / floats). 136 halves = 272B row stride ->
// 16-lane fragment reads hit distinct banks; rows stay 16B aligned.
#define XLD 136
#define WLD 136
#define QLD 136
#define KLD 136
#define VTLD 72
#define SLD 68   // floats
#define PLD 72
#define OLD 136
#define PWLD 136

#define OFF_X  0                 // 64*136*2  = 17408   (reused as O later)
#define OFF_W  17408             // 384*136*2 = 104448  (reused as S + projW)
#define OFF_S  17408             // 4*64*68*4 = 69632
#define OFF_PW 87040             // 128*136*2 = 34816
#define OFF_Q  121856            // 17408
#define OFF_K  139264            // 17408
#define OFF_VT 156672            // 128*72*2  = 18432
#define OFF_P  175104            // 4*64*72*2 = 36864
#define SMEM_BYTES 211968

#define WMMA_F16(A, B, Cacc) \
    __builtin_amdgcn_wmma_f32_16x16x32_f16(false, (A), false, (B), (short)0, (Cacc), false, false)

// ---- WMMA fragment loaders (V_WMMA_F32_16X16X32_F16 layouts) ----
// A (16x32, MxK): lane m = lane&15, g = lane>>4.
//   e<8 : K = g*8 + e ; e>=8 : K = 16 + g*8 + (e-8)  -> two contiguous 8-half runs.
__device__ __forceinline__ v16h load_frag_A(const _Float16* base, int ld, int m0, int k0) {
    const int lane = threadIdx.x & 31;
    const int g = lane >> 4;
    const _Float16* row = base + (m0 + (lane & 15)) * ld + k0 + g * 8;
    h8 lo = *(const h8*)(row);
    h8 hi = *(const h8*)(row + 16);
    return __builtin_shufflevector(lo, hi, 0,1,2,3,4,5,6,7,8,9,10,11,12,13,14,15);
}
// B (32x16, KxN) from a row-major array stored as [N][K]:
//   lane n = lane&15, e<8 : K = g*8 + e ; e>=8 : K = 16 + g*8 + (e-8) (same as A).
__device__ __forceinline__ v16h load_frag_B(const _Float16* base, int ld, int n0, int k0) {
    const int lane = threadIdx.x & 31;
    const int g = lane >> 4;
    const _Float16* row = base + (n0 + (lane & 15)) * ld + k0 + g * 8;
    h8 lo = *(const h8*)(row);
    h8 hi = *(const h8*)(row + 16);
    return __builtin_shufflevector(lo, hi, 0,1,2,3,4,5,6,7,8,9,10,11,12,13,14,15);
}

__device__ __forceinline__ int regid(int p) { return p < 49 ? 0 : (p < 53 ? 1 : 2); }

extern "C" __global__ __launch_bounds__(256)
void swin_window_attn_kernel(const float* __restrict__ x,
                             const float* __restrict__ qkv_w,
                             const float* __restrict__ qkv_b,
                             const float* __restrict__ proj_w,
                             const float* __restrict__ proj_b,
                             const float* __restrict__ rpt,
                             float* __restrict__ out) {
    extern __shared__ __align__(16) char smem[];
    _Float16* Xs  = (_Float16*)(smem + OFF_X);
    _Float16* Wqs = (_Float16*)(smem + OFF_W);
    float*    Ss  = (float*)   (smem + OFF_S);
    _Float16* PWs = (_Float16*)(smem + OFF_PW);
    _Float16* Qs  = (_Float16*)(smem + OFF_Q);
    _Float16* Ks  = (_Float16*)(smem + OFF_K);
    _Float16* Vt  = (_Float16*)(smem + OFF_VT);
    _Float16* Ps  = (_Float16*)(smem + OFF_P);
    _Float16* Os  = (_Float16*)(smem + OFF_X);   // overlays X (dead after GEMM1)

    const int tid  = threadIdx.x;
    // wave id is wave-uniform: force it into an SGPR so all job indices,
    // part selectors and LDS tile bases scalarize (s_cbranch, not v_cmpx).
    const int wave = __builtin_amdgcn_readfirstlane(tid >> 5);
    const int lane = tid & 31;
    const int ln   = lane & 15;
    const int lg   = lane >> 4;

    const int wid = blockIdx.x;      // window id: b*64 + w
    const int b   = wid >> 6;
    const int w   = wid & 63;
    const int wh  = w >> 3;
    const int ww  = w & 7;

    // ---- phase 1: stage shifted input window + qkv weights (f32 -> f16) ----
    for (int it = tid; it < NPAD * (CC / 4); it += 256) {
        int r = it >> 5, c4 = it & 31;
        h4 hv = {0, 0, 0, 0};
        if (r < NTOK) {
            int i = r / 7, j = r % 7;
            int hs = (wh * 7 + i + 3) % 56;
            int vs = (ww * 7 + j + 3) % 56;
            const float4 f = *(const float4*)(x + (((size_t)b * 56 + hs) * 56 + vs) * CC + c4 * 4);
            hv = h4{(_Float16)f.x, (_Float16)f.y, (_Float16)f.z, (_Float16)f.w};
        }
        *(h4*)(Xs + r * XLD + c4 * 4) = hv;
    }
    for (int it = tid; it < 384 * (CC / 4); it += 256) {
        int r = it >> 5, c4 = it & 31;
        const float4 f = *(const float4*)(qkv_w + (size_t)r * CC + c4 * 4);
        *(h4*)(Wqs + r * WLD + c4 * 4) = h4{(_Float16)f.x, (_Float16)f.y, (_Float16)f.z, (_Float16)f.w};
    }
    __syncthreads();

    // ---- phase 2: QKV GEMM  [64x128] x [128x384]^T ----
    // mt is constant per wave -> hoist the 4 A k-fragments; nt = (wave>>2) + 2*i.
    // Two independent accumulator chains let the scheduler keep two B
    // fragments in flight instead of a full dscnt-0 wait per WMMA.
    {
        const int mt = wave & 3;
        v16h a0 = load_frag_A(Xs, XLD, mt * 16, 0);
        v16h a1 = load_frag_A(Xs, XLD, mt * 16, 32);
        v16h a2 = load_frag_A(Xs, XLD, mt * 16, 64);
        v16h a3 = load_frag_A(Xs, XLD, mt * 16, 96);
#pragma unroll 1
        for (int i = 0; i < 12; ++i) {
            const int nt = (wave >> 2) + 2 * i;
            v16h b0 = load_frag_B(Wqs, WLD, nt * 16, 0);
            v16h b1 = load_frag_B(Wqs, WLD, nt * 16, 32);
            v16h b2 = load_frag_B(Wqs, WLD, nt * 16, 64);
            v16h b3 = load_frag_B(Wqs, WLD, nt * 16, 96);
            v8f acc0 = {}, acc1 = {};
            acc0 = WMMA_F16(a0, b0, acc0);
            acc1 = WMMA_F16(a1, b1, acc1);
            acc0 = WMMA_F16(a2, b2, acc0);
            acc1 = WMMA_F16(a3, b3, acc1);

            const int part = nt >> 3;               // scalar: 0=q 1=k 2=v
            const int oc   = (nt & 7) * 16 + ln;
            const float bias = qkv_b[part * CC + oc];
            if (part == 0) {
#pragma unroll
                for (int r = 0; r < 8; ++r)
                    Qs[(mt * 16 + lg * 8 + r) * QLD + oc] =
                        (_Float16)((acc0[r] + acc1[r] + bias) * 0.17677669529663689f);
            } else if (part == 1) {
#pragma unroll
                for (int r = 0; r < 8; ++r)
                    Ks[(mt * 16 + lg * 8 + r) * KLD + oc] = (_Float16)(acc0[r] + acc1[r] + bias);
            } else {
#pragma unroll
                for (int r = 0; r < 8; ++r)
                    Vt[oc * VTLD + (mt * 16 + lg * 8 + r)] = (_Float16)(acc0[r] + acc1[r] + bias);
            }
        }
    }
    __syncthreads();

    // ---- phase 3: stage proj weights (into old W region) + Q*K^T per head ----
    for (int it = tid; it < CC * (CC / 4); it += 256) {
        int r = it >> 5, c4 = it & 31;
        const float4 f = *(const float4*)(proj_w + (size_t)r * CC + c4 * 4);
        *(h4*)(PWs + r * PWLD + c4 * 4) = h4{(_Float16)f.x, (_Float16)f.y, (_Float16)f.z, (_Float16)f.w};
    }
#pragma unroll 1
    for (int i = 0; i < 8; ++i) {
        const int job = wave + 8 * i;
        const int h = job >> 4, mt = (job >> 2) & 3, nt = job & 3;
        v16h a  = load_frag_A(Qs, QLD, mt * 16, h * 32);        // head dim 32 = 1 K-step
        v16h bm = load_frag_B(Ks, KLD, nt * 16, h * 32);
        v8f acc = {};
        acc = WMMA_F16(a, bm, acc);
        float* Sh = Ss + (size_t)h * 64 * SLD;
#pragma unroll
        for (int r = 0; r < 8; ++r)
            Sh[(mt * 16 + lg * 8 + r) * SLD + nt * 16 + ln] = acc[r];
    }
    __syncthreads();

    // ---- phase 4: bias + shift-mask + softmax; one thread per (head, row) ----
    {
        int h = tid >> 6, m = tid & 63;
        float*    Sh = Ss + ((size_t)h * 64 + m) * SLD;
        _Float16* Ph = Ps + ((size_t)h * 64 + m) * PLD;
        if (m < NTOK) {
            int mi = m / 7, mj = m % 7;
            int idm = regid(wh * 7 + mi) * 3 + regid(ww * 7 + mj);
            float mx = -1e30f;
            for (int n = 0; n < NTOK; ++n) {
                int ni = n / 7, nj = n % 7;
                int ridx = (mi - ni + 6) * 13 + (mj - nj + 6);
                int idn  = regid(wh * 7 + ni) * 3 + regid(ww * 7 + nj);
                float s = Sh[n] + rpt[ridx * 4 + h] + ((idm != idn) ? -100.0f : 0.0f);
                Sh[n] = s;
                mx = fmaxf(mx, s);
            }
            float sum = 0.0f;
            for (int n = 0; n < NTOK; ++n) { float e = __expf(Sh[n] - mx); sum += e; Sh[n] = e; }
            float inv = 1.0f / sum;
            for (int n = 0; n < NTOK; ++n) Ph[n] = (_Float16)(Sh[n] * inv);
            for (int n = NTOK; n < NPAD; ++n) Ph[n] = (_Float16)0.0f;
        } else {
            for (int n = 0; n < NPAD; ++n) Ph[n] = (_Float16)0.0f;
        }
    }
    __syncthreads();

    // ---- phase 5: attn @ V -> O (f16, overlays X region) ----
    // (mt, dt) constant per wave; loop over heads; two independent chains.
    {
        const int mt = (wave >> 1) & 3;
        const int dt = wave & 1;
#pragma unroll 1
        for (int h = 0; h < 4; ++h) {
            v16h a0 = load_frag_A(Ps + (size_t)h * 64 * PLD, PLD, mt * 16, 0);
            v16h a1 = load_frag_A(Ps + (size_t)h * 64 * PLD, PLD, mt * 16, 32);
            v16h b0 = load_frag_B(Vt, VTLD, h * 32 + dt * 16, 0);   // Vt: [channel][token]
            v16h b1 = load_frag_B(Vt, VTLD, h * 32 + dt * 16, 32);
            v8f acc0 = {}, acc1 = {};
            acc0 = WMMA_F16(a0, b0, acc0);
            acc1 = WMMA_F16(a1, b1, acc1);
            const int col = h * 32 + dt * 16 + ln;
#pragma unroll
            for (int r = 0; r < 8; ++r)
                Os[(mt * 16 + lg * 8 + r) * OLD + col] = (_Float16)(acc0[r] + acc1[r]);
        }
    }
    __syncthreads();

    // ---- phase 6: output projection + reverse-shift scatter to global ----
    // nt is constant per wave -> hoist the 4 B k-fragments; loop over mt.
    {
        const int nt = wave;
        v16h b0 = load_frag_B(PWs, PWLD, nt * 16, 0);
        v16h b1 = load_frag_B(PWs, PWLD, nt * 16, 32);
        v16h b2 = load_frag_B(PWs, PWLD, nt * 16, 64);
        v16h b3 = load_frag_B(PWs, PWLD, nt * 16, 96);
        const int oc = nt * 16 + ln;
        const float bias = proj_b[oc];
#pragma unroll 1
        for (int mt = 0; mt < 4; ++mt) {
            v16h a0 = load_frag_A(Os, OLD, mt * 16, 0);
            v16h a1 = load_frag_A(Os, OLD, mt * 16, 32);
            v16h a2 = load_frag_A(Os, OLD, mt * 16, 64);
            v16h a3 = load_frag_A(Os, OLD, mt * 16, 96);
            v8f acc0 = {}, acc1 = {};
            acc0 = WMMA_F16(a0, b0, acc0);
            acc1 = WMMA_F16(a1, b1, acc1);
            acc0 = WMMA_F16(a2, b2, acc0);
            acc1 = WMMA_F16(a3, b3, acc1);
#pragma unroll
            for (int r = 0; r < 8; ++r) {
                int m = mt * 16 + lg * 8 + r;
                if (m < NTOK) {
                    int i = m / 7, j = m % 7;
                    int dh = (wh * 7 + i + 3) % 56;
                    int dw = (ww * 7 + j + 3) % 56;
                    out[(((size_t)b * 56 + dh) * 56 + dw) * CC + oc] = acc0[r] + acc1[r] + bias;
                }
            }
        }
    }
}

extern "C" void kernel_launch(void* const* d_in, const int* in_sizes, int n_in,
                              void* d_out, int out_size, void* d_ws, size_t ws_size,
                              hipStream_t stream) {
    const float* x      = (const float*)d_in[0];
    const float* qkv_w  = (const float*)d_in[1];
    const float* qkv_b  = (const float*)d_in[2];
    const float* proj_w = (const float*)d_in[3];
    const float* proj_b = (const float*)d_in[4];
    const float* rpt    = (const float*)d_in[5];
    float* o = (float*)d_out;

    (void)in_sizes; (void)n_in; (void)out_size; (void)d_ws; (void)ws_size;

    hipFuncSetAttribute((const void*)swin_window_attn_kernel,
                        hipFuncAttributeMaxDynamicSharedMemorySize, SMEM_BYTES);
    // one block per window: 64 batches * 64 windows = 4096 blocks, 8 waves each
    swin_window_attn_kernel<<<dim3(64 * 64), dim3(256), SMEM_BYTES, stream>>>(
        x, qkv_w, qkv_b, proj_w, proj_b, rpt, o);
}
